// MambaLM_43284680409478
// MI455X (gfx1250) — matrix operand order repaired and checked by reading, not compiled, not run on hardware
//
#include <hip/hip_runtime.h>

#define B_   2
#define L_   2048
#define D_   768
#define DI_  1536
#define DS_  16
#define DTR_ 48
#define V_   32000
#define NL_  4
#define NC_  16            // scan chunks
#define LC_  (L_ / NC_)    // 128 timesteps per chunk
#define LDT  40            // LDS tile row stride (bf16 elems): 80B = 64B data + 16B TDM pad

typedef __bf16 bf16;
typedef __attribute__((ext_vector_type(16))) __bf16 bf16x16;
typedef __attribute__((ext_vector_type(8)))  float  f32x8;
typedef __attribute__((ext_vector_type(4)))  unsigned int u32x4;
typedef __attribute__((ext_vector_type(8)))  int    i32x8;
typedef __attribute__((ext_vector_type(4)))  int    i32x4;

// ---------------------------------------------------------------- WMMA core
__device__ inline f32x8 wmma_bf16(bf16x16 a, bf16x16 b, f32x8 c) {
  // D = A(16x32) * B(32x16) + C, f32 accumulate
  return __builtin_amdgcn_wmma_f32_16x16x32_bf16(false, a, false, b, (short)0, c, false, false);
}

// A fragment (16x32 bf16, ISA layout): lanes 0-15 row m=lane, elems[0..7]=K0..7,
// elems[8..15]=K16..23 ; lanes 16-31 row m=lane-16, K8..15 / K24..31.
__device__ inline bf16x16 lds_frag_a(const bf16* base, int lane) {
  int r = lane & 15;
  int off = (lane & 16) ? 8 : 0;
  const bf16* p = base + r * LDT + off;
  bf16x16 f;
  ((uint4*)&f)[0] = *(const uint4*)(p);
  ((uint4*)&f)[1] = *(const uint4*)(p + 16);
  return f;
}

// B fragment (32x16 bf16): lanes 0-15 col n=lane, K0..7/K16..23 ; lanes 16-31 col n=lane-16, K8..15/K24..31.
__device__ inline bf16x16 lds_frag_b(const bf16* base, int lane) {
  int r = lane & 15;
  int off = (lane & 16) ? 8 : 0;
  const bf16* p = base + r * LDT + off;
  bf16x16 f;
  ((uint4*)&f)[0] = ((const uint4*)p)[0];
  ((uint4*)&f)[1] = *(const uint4*)(p + 16);
  return f;
}

// low 32 bits of a generic LDS pointer == LDS byte offset (aperture in high bits)
__device__ inline unsigned int lds_byte_off(const void* p) {
  return (unsigned int)(unsigned long long)(uintptr_t)p;
}

// -------------------------------------------------- TDM 2-D tile -> LDS
// Loads tile_rows x 32 bf16 tile starting at gptr (row stride = row_stride elems).
// Zero-fills past (cols_rem x rows_rem). TDM pads each 64B row with 16B in LDS
// (pad_interval=16 DWORDs, pad_amount=4 DWORDs) -> LDT=40 elem row stride.
__device__ inline void tdm_load_tile(unsigned int lds_off, const bf16* gptr,
                                     int rows_rem, int cols_rem, int tile_rows,
                                     long row_stride) {
  unsigned long long ga = (unsigned long long)(uintptr_t)gptr;
  unsigned int td0 = (unsigned int)(cols_rem < 0 ? 0 : cols_rem);
  unsigned int td1 = (unsigned int)(rows_rem < 0 ? 0 : rows_rem);
  unsigned long long s0 = (unsigned long long)row_stride;

  u32x4 g0;
  g0[0] = 1u;                                             // count=1, user descriptor
  g0[1] = lds_off;                                        // LDS byte address
  g0[2] = (unsigned int)ga;                               // global_addr[31:0]
  g0[3] = (unsigned int)((ga >> 32) & 0x01FFFFFFu)        // global_addr[56:32]
        | 0x80000000u;                                    // type=2 (image) [127:126]

  i32x8 g1;
  g1[0] = (int)((1u << 16)                                // data_size: 2 bytes
              | (1u << 20)                                // pad_enable
              | (3u << 22)                                // pad_interval: 16 DWORDs
              | (3u << 25));                              // pad_amount: 4 DWORDs
  g1[1] = (int)((td0 & 0xFFFFu) << 16);                   // [31:16] tensor_dim0 lo
  g1[2] = (int)((td0 >> 16) | ((td1 & 0xFFFFu) << 16));   // dim0 hi | dim1 lo
  g1[3] = (int)((td1 >> 16) | (32u << 16));               // dim1 hi | tile_dim0=32
  g1[4] = (int)(unsigned int)tile_rows;                   // tile_dim1 | tile_dim2=0
  g1[5] = (int)(unsigned int)(s0 & 0xFFFFFFFFull);        // tensor_dim0_stride lo
  g1[6] = (int)(unsigned int)((s0 >> 32) & 0xFFFFull);    // stride0 hi | stride1 lo=0
  g1[7] = 0;
  i32x4 gz4  = {0, 0, 0, 0};
  i32x8 gz8  = {0, 0, 0, 0, 0, 0, 0, 0};
  // clang-23 / therock form: (g0, g1, g2, g3, extra, cpol)
  __builtin_amdgcn_tensor_load_to_lds(g0, g1, gz4, gz4, gz8, 0);
}

// ------------------------------------------------------------- GEMM kernel
// C(M,N) f32 = A(M,K) bf16 row-major  x  W(N,K)^T bf16 row-major
// EPI 0: C = acc
// EPI 1: C = softplus(acc + bias[n])           (dt projection)
// EPI 2: C = resid + scale_p[scale_i] * acc    (out projection + residual)
template<int EPI>
__global__ __launch_bounds__(256) void gemm_bf16_kernel(
    const bf16* __restrict__ A, const bf16* __restrict__ W,
    float* __restrict__ C, int M, int N, int K,
    const float* __restrict__ bias, const float* __restrict__ resid,
    const float* __restrict__ scale_p, int scale_i)
{
  __shared__ __attribute__((aligned(16))) bf16 As[2][128 * LDT];
  __shared__ __attribute__((aligned(16))) bf16 Bs[2][64 * LDT];

  int tid  = threadIdx.x;
  int lane = tid & 31;
  int w    = tid >> 5;      // 8 waves
  int wm   = w >> 1;        // 0..3  -> 32-row slab
  int wn   = w & 1;         // 0..1  -> 32-col slab
  int m0   = blockIdx.y * 128;
  int n0   = blockIdx.x * 64;

  f32x8 acc00 = {}, acc01 = {}, acc10 = {}, acc11 = {};

  // wave 0 drives the Tensor Data Mover; double-buffered LDS staging
  if (tid < 32) {
    tdm_load_tile(lds_byte_off(&As[0][0]), A + (size_t)m0 * K,       M - m0, K, 128, K);
    tdm_load_tile(lds_byte_off(&Bs[0][0]), W + (size_t)n0 * K,       N - n0, K, 64,  K);
  }

  int buf = 0;
  for (int kc = 0; kc < K; kc += 32, buf ^= 1) {
    if (tid < 32) {
      if (kc + 32 < K) {
        tdm_load_tile(lds_byte_off(&As[buf ^ 1][0]), A + (size_t)m0 * K + kc + 32,
                      M - m0, K - (kc + 32), 128, K);
        tdm_load_tile(lds_byte_off(&Bs[buf ^ 1][0]), W + (size_t)n0 * K + kc + 32,
                      N - n0, K - (kc + 32), 64, K);
        __builtin_amdgcn_s_wait_tensorcnt(2);  // in-order: current chunk's 2 TDMs done
      } else {
        __builtin_amdgcn_s_wait_tensorcnt(0);
      }
    }
    __syncthreads();   // release compute on buf

    bf16x16 a0 = lds_frag_a(&As[buf][(wm * 32 + 0)  * LDT], lane);
    bf16x16 a1 = lds_frag_a(&As[buf][(wm * 32 + 16) * LDT], lane);
    bf16x16 b0 = lds_frag_b(&Bs[buf][(wn * 32 + 0)  * LDT], lane);
    bf16x16 b1 = lds_frag_b(&Bs[buf][(wn * 32 + 16) * LDT], lane);

    acc00 = wmma_bf16(a0, b0, acc00);
    acc01 = wmma_bf16(a0, b1, acc01);
    acc10 = wmma_bf16(a1, b0, acc10);
    acc11 = wmma_bf16(a1, b1, acc11);
    __syncthreads();   // buf reusable for TDM next+1
  }

  float scale = 1.f;
  if (EPI == 2) scale = scale_p[scale_i];
  int ro = (lane & 16) ? 8 : 0;   // C/D layout: lanes>=16 hold rows j+8
  int cl = lane & 15;

  auto store_tile = [&](const f32x8& a, int mt, int nt) {
    int gm0 = m0 + wm * 32 + mt * 16 + ro;
    int gn  = n0 + wn * 32 + nt * 16 + cl;
    if (gn >= N) return;
#pragma unroll
    for (int j = 0; j < 8; j++) {
      int gm = gm0 + j;
      size_t idx = (size_t)gm * N + gn;
      float v = a[j];
      if (EPI == 1) {
        v += bias[gn];
        v = (v > 20.f) ? v : log1pf(__expf(v));
      } else if (EPI == 2) {
        v = resid[idx] + scale * v;
      }
      C[idx] = v;
    }
  };
  store_tile(acc00, 0, 0);
  store_tile(acc01, 0, 1);
  store_tile(acc10, 1, 0);
  store_tile(acc11, 1, 1);
}

// ------------------------------------------------------------- helpers
__global__ void cvt_kernel(const float* __restrict__ in, bf16* __restrict__ out, long n) {
  long i = (long)blockIdx.x * blockDim.x + threadIdx.x;
  long stride = (long)gridDim.x * blockDim.x;
  for (; i < n; i += stride) out[i] = (bf16)in[i];
}

__global__ void embed_kernel(const int* __restrict__ tokens, const float* __restrict__ emb,
                             float* __restrict__ x) {
  int i = blockIdx.x * 256 + threadIdx.x;
  if (i >= B_ * L_ * D_) return;
  int row = i / D_, col = i - row * D_;
  x[i] = emb[(size_t)tokens[row] * D_ + col];
}

__global__ void layernorm_kernel(const float* __restrict__ x, const float* __restrict__ w,
                                 const float* __restrict__ b, bf16* __restrict__ out) {
  int row = blockIdx.x;
  int tid = threadIdx.x;
  const float* xr = x + (size_t)row * D_;
  float s = 0.f, q = 0.f;
  for (int c = tid; c < D_; c += 256) { float v = xr[c]; s += v; q += v * v; }
  for (int o = 16; o > 0; o >>= 1) { s += __shfl_xor(s, o, 32); q += __shfl_xor(q, o, 32); }
  __shared__ float rs[8], rq[8];
  int wid = tid >> 5, lane = tid & 31;
  if (lane == 0) { rs[wid] = s; rq[wid] = q; }
  __syncthreads();
  if (tid == 0) {
    float ts = 0.f, tq = 0.f;
    for (int i = 0; i < 8; i++) { ts += rs[i]; tq += rq[i]; }
    rs[0] = ts; rq[0] = tq;
  }
  __syncthreads();
  float mean = rs[0] * (1.f / D_);
  float var  = rq[0] * (1.f / D_) - mean * mean;
  float rstd = rsqrtf(var + 1e-5f);
  for (int c = tid; c < D_; c += 256)
    out[(size_t)row * D_ + c] = (bf16)((xr[c] - mean) * rstd * w[c] + b[c]);
}

// causal depthwise conv (DC=4) + bias + SiLU ; input = first DI_ cols of xz
__global__ void conv_silu_kernel(const float* __restrict__ xz, const float* __restrict__ cw,
                                 const float* __restrict__ cb, float* __restrict__ uf,
                                 bf16* __restrict__ ub) {
  int idx = blockIdx.x * 256 + threadIdx.x;
  if (idx >= B_ * L_ * DI_) return;
  int d  = idx % DI_;
  int bl = idx / DI_;
  int t  = bl % L_;
  int b  = bl / L_;
  float acc = cb[d];
#pragma unroll
  for (int j = 0; j < 4; j++) {
    int tt = t + j - 3;
    if (tt >= 0) acc += cw[d * 4 + j] * xz[((size_t)(b * L_ + tt)) * (2 * DI_) + d];
  }
  float sv = acc / (1.f + __expf(-acc));
  uf[idx] = sv;
  ub[idx] = (bf16)sv;
}

// first 48 cols of x_dbl (stride 80) -> bf16 (stride 48) for dt GEMM
__global__ void cvt48_kernel(const float* __restrict__ xdbl, bf16* __restrict__ out) {
  int i = blockIdx.x * 256 + threadIdx.x;
  if (i >= B_ * L_ * DTR_) return;
  int m = i / DTR_, c = i - m * DTR_;
  out[i] = (bf16)xdbl[(size_t)m * 80 + c];
}

// ---------------------------- chunked selective scan (affine recurrence) ----
// PHASE 1: per chunk, h from 0 -> store end-state S_j and decay product P_j
// PHASE 2: per chunk, h from corrected init -> emit gated y (bf16)
template<int PHASE>
__global__ __launch_bounds__(256) void scan_kernel(
    const float* __restrict__ dtf, const float* __restrict__ uf,
    const float* __restrict__ xz,  const float* __restrict__ xdbl,
    const float* __restrict__ A_log, const float* __restrict__ Dp,
    float* __restrict__ He, float* __restrict__ Pb,
    const float* __restrict__ Hi, bf16* __restrict__ yb)
{
  int d = blockIdx.x * 256 + threadIdx.x;
  int b = blockIdx.y;
  int j = blockIdx.z;
  float Af[DS_];
#pragma unroll
  for (int s = 0; s < DS_; s++) Af[s] = -__expf(A_log[(size_t)d * DS_ + s]);
  float h[DS_], P[DS_];
  size_t sbase = ((size_t)(j * B_ + b) * DI_ + d) * DS_;
  if (PHASE == 1) {
#pragma unroll
    for (int s = 0; s < DS_; s++) { h[s] = 0.f; P[s] = 1.f; }
  } else {
#pragma unroll
    for (int s = 0; s < DS_; s++) h[s] = Hi[sbase + s];
  }
  float Dpd = Dp[d];
  int t0 = j * LC_, t1 = t0 + LC_;
  for (int t = t0; t < t1; t++) {
    size_t rb = (size_t)(b * L_ + t);
    float dt = dtf[rb * DI_ + d];
    float uu = uf[rb * DI_ + d];
    float du = dt * uu;
    const float* xr = xdbl + rb * (DTR_ + 2 * DS_);
    float y = 0.f;
#pragma unroll
    for (int s = 0; s < DS_; s++) {
      float e = __expf(dt * Af[s]);
      h[s] = e * h[s] + du * xr[DTR_ + s];
      if (PHASE == 1) P[s] *= e;
      else            y += h[s] * xr[DTR_ + DS_ + s];
    }
    if (PHASE == 2) {
      float zz = xz[rb * (2 * DI_) + DI_ + d];
      y += uu * Dpd;
      y *= zz / (1.f + __expf(-zz));
      yb[rb * DI_ + d] = (bf16)y;
    }
  }
  if (PHASE == 1) {
#pragma unroll
    for (int s = 0; s < DS_; s++) { He[sbase + s] = h[s]; Pb[sbase + s] = P[s]; }
  }
}

// chain chunk boundaries: h_init[j+1] = P_j * h_init[j] + S_j
__global__ void scan_mid_kernel(const float* __restrict__ He, const float* __restrict__ Pb,
                                float* __restrict__ Hi) {
  int d = blockIdx.x * 256 + threadIdx.x;
  int b = blockIdx.y;
  if (d >= DI_) return;
  float h[DS_];
#pragma unroll
  for (int s = 0; s < DS_; s++) h[s] = 0.f;
  for (int j = 0; j < NC_; j++) {
    size_t base = ((size_t)(j * B_ + b) * DI_ + d) * DS_;
#pragma unroll
    for (int s = 0; s < DS_; s++) {
      Hi[base + s] = h[s];
      h[s] = Pb[base + s] * h[s] + He[base + s];
    }
  }
}

// --------------------------------------------------------------- launcher
extern "C" void kernel_launch(void* const* d_in, const int* in_sizes, int n_in,
                              void* d_out, int out_size, void* d_ws, size_t ws_size,
                              hipStream_t stream) {
  const int*   tokens = (const int*)  d_in[0];
  const float* emb    = (const float*)d_in[1];
  const float* ln_w   = (const float*)d_in[2];
  const float* ln_b   = (const float*)d_in[3];
  const float* in_w   = (const float*)d_in[4];
  const float* conv_w = (const float*)d_in[5];
  const float* conv_b = (const float*)d_in[6];
  const float* xp_w   = (const float*)d_in[7];
  const float* dt_w   = (const float*)d_in[8];
  const float* dt_b   = (const float*)d_in[9];
  const float* A_log  = (const float*)d_in[10];
  const float* Dp     = (const float*)d_in[11];
  const float* out_w  = (const float*)d_in[12];
  const float* res_s  = (const float*)d_in[13];
  const float* fin_w  = (const float*)d_in[14];
  const float* fin_b  = (const float*)d_in[15];
  const float* head_w = (const float*)d_in[16];
  float* out = (float*)d_out;

  const size_t BL = (size_t)B_ * L_;         // 4096 rows
  char* cur = (char*)d_ws;
  auto alloc = [&](size_t bytes) -> void* {
    void* r = (void*)cur;
    cur += (bytes + 255) & ~(size_t)255;
    return r;
  };

  bf16* wb_in   = (bf16*) alloc((size_t)NL_ * 2 * DI_ * D_ * 2);
  bf16* wb_x    = (bf16*) alloc((size_t)NL_ * (DTR_ + 2 * DS_) * DI_ * 2);
  bf16* wb_dt   = (bf16*) alloc((size_t)NL_ * DI_ * DTR_ * 2);
  bf16* wb_out  = (bf16*) alloc((size_t)NL_ * D_ * DI_ * 2);
  bf16* wb_head = (bf16*) alloc((size_t)V_ * D_ * 2);
  float* xf     = (float*)alloc(BL * D_ * 4);
  bf16*  hb     = (bf16*) alloc(BL * D_ * 2);
  float* xz     = (float*)alloc(BL * 2 * DI_ * 4);
  float* uf     = (float*)alloc(BL * DI_ * 4);
  bf16*  ub     = (bf16*) alloc(BL * DI_ * 2);
  float* xdbl   = (float*)alloc(BL * (DTR_ + 2 * DS_) * 4);
  bf16*  xdbl48 = (bf16*) alloc(BL * DTR_ * 2);
  float* dtf    = (float*)alloc(BL * DI_ * 4);
  bf16*  yb     = (bf16*) alloc(BL * DI_ * 2);
  float* He     = (float*)alloc((size_t)NC_ * B_ * DI_ * DS_ * 4);
  float* Pb     = (float*)alloc((size_t)NC_ * B_ * DI_ * DS_ * 4);
  float* Hi     = (float*)alloc((size_t)NC_ * B_ * DI_ * DS_ * 4);

  // weights -> bf16 (once per call; deterministic)
  cvt_kernel<<<1024, 256, 0, stream>>>(in_w,   wb_in,   (long)NL_ * 2 * DI_ * D_);
  cvt_kernel<<<256,  256, 0, stream>>>(xp_w,   wb_x,    (long)NL_ * (DTR_ + 2 * DS_) * DI_);
  cvt_kernel<<<256,  256, 0, stream>>>(dt_w,   wb_dt,   (long)NL_ * DI_ * DTR_);
  cvt_kernel<<<1024, 256, 0, stream>>>(out_w,  wb_out,  (long)NL_ * D_ * DI_);
  cvt_kernel<<<2048, 256, 0, stream>>>(head_w, wb_head, (long)V_ * D_);

  embed_kernel<<<(int)((BL * D_ + 255) / 256), 256, 0, stream>>>(tokens, emb, xf);

  for (int l = 0; l < NL_; l++) {
    layernorm_kernel<<<(int)BL, 256, 0, stream>>>(xf, ln_w + l * D_, ln_b + l * D_, hb);

    // xz = h @ in_w^T : M=4096, N=3072, K=768
    gemm_bf16_kernel<0><<<dim3((2 * DI_ + 63) / 64, (int)(BL / 128)), 256, 0, stream>>>(
        hb, wb_in + (size_t)l * 2 * DI_ * D_, xz, (int)BL, 2 * DI_, D_,
        nullptr, nullptr, nullptr, 0);

    conv_silu_kernel<<<(int)((BL * DI_ + 255) / 256), 256, 0, stream>>>(
        xz, conv_w + (size_t)l * DI_ * 4, conv_b + (size_t)l * DI_, uf, ub);

    // x_dbl = u @ xp_w^T : M=4096, N=80, K=1536
    gemm_bf16_kernel<0><<<dim3(((DTR_ + 2 * DS_) + 63) / 64, (int)(BL / 128)), 256, 0, stream>>>(
        ub, wb_x + (size_t)l * (DTR_ + 2 * DS_) * DI_, xdbl, (int)BL, DTR_ + 2 * DS_, DI_,
        nullptr, nullptr, nullptr, 0);

    cvt48_kernel<<<(int)((BL * DTR_ + 255) / 256), 256, 0, stream>>>(xdbl, xdbl48);

    // dt = softplus(xdbl[:,:48] @ dt_w^T + dt_b) : M=4096, N=1536, K=48
    gemm_bf16_kernel<1><<<dim3(DI_ / 64, (int)(BL / 128)), 256, 0, stream>>>(
        xdbl48, wb_dt + (size_t)l * DI_ * DTR_, dtf, (int)BL, DI_, DTR_,
        dt_b + (size_t)l * DI_, nullptr, nullptr, 0);

    // chunked selective scan
    scan_kernel<1><<<dim3(DI_ / 256, B_, NC_), 256, 0, stream>>>(
        dtf, uf, xz, xdbl, A_log + (size_t)l * DI_ * DS_, Dp + (size_t)l * DI_,
        He, Pb, nullptr, nullptr);
    scan_mid_kernel<<<dim3(DI_ / 256, B_), 256, 0, stream>>>(He, Pb, Hi);
    scan_kernel<2><<<dim3(DI_ / 256, B_, NC_), 256, 0, stream>>>(
        dtf, uf, xz, xdbl, A_log + (size_t)l * DI_ * DS_, Dp + (size_t)l * DI_,
        nullptr, nullptr, Hi, yb);

    // x += res_scale[l] * (y @ out_w^T) : M=4096, N=768, K=1536
    gemm_bf16_kernel<2><<<dim3(D_ / 64, (int)(BL / 128)), 256, 0, stream>>>(
        yb, wb_out + (size_t)l * D_ * DI_, xf, (int)BL, D_, DI_,
        nullptr, xf, res_s, l);
  }

  layernorm_kernel<<<(int)BL, 256, 0, stream>>>(xf, fin_w, fin_b, hb);

  // logits = x @ head_w^T : M=4096, N=32000, K=768
  gemm_bf16_kernel<0><<<dim3((V_ + 63) / 64, (int)(BL / 128)), 256, 0, stream>>>(
      hb, wb_head, out, (int)BL, V_, D_, nullptr, nullptr, nullptr, 0);
}
